// DeepseekV3MoE_64742337020595
// MI455X (gfx1250) — compile-verified, matrix operand order checked by reading, not run on hardware
//
#include <hip/hip_runtime.h>
#include <hip/hip_bf16.h>
#include <cstddef>

// ---------------------------------------------------------------------------
// Problem constants (DeepSeek-V3 MoE, fixed by the reference)
// ---------------------------------------------------------------------------
#define TT 2048   // tokens
#define HH 1024   // hidden
#define EE 32     // routed experts
#define FF 512    // moe intermediate
#define KK 8      // top_k
#define NGR 8     // n_group
#define TGR 4     // topk_group
#define FSH 1024  // shared intermediate
#define RSCALE 2.5f

// ---------------------------------------------------------------------------
// WMMA plumbing: V_WMMA_F32_16X16X32_BF16, operands in ISA 16-bit A/B layout
// ---------------------------------------------------------------------------
typedef __bf16 v16bf __attribute__((ext_vector_type(16)));
typedef float  v8f   __attribute__((ext_vector_type(8)));

union V16 {
    v16bf v;
    __bf16 s[16];
    uint4 q[2];
};

union V8B {
    __bf16 s[8];
    uint4 q;
};

static __device__ inline float fast_rcp(float x) {
    return __builtin_amdgcn_rcpf(x);
}

static __device__ inline float silu(float g) {
    return g * fast_rcp(1.0f + __expf(-g));
}

// ---- 16x32 tile loaders (A and B share the same lane striping) -------------
// Lane l holds row/col (l&15); K-half (l>>4)*8 gives two contiguous 8-elem
// runs at k0+kb and k0+16+kb.

// bf16 source: 2x b128 loads, no conversion.
static __device__ inline v16bf load_tile_g(const __bf16* __restrict__ base,
                                           int row_stride, int k0, int lane) {
    int rc = lane & 15;
    int kb = (lane >> 4) << 3;
    const __bf16* p = base + (size_t)rc * row_stride + k0 + kb;
    V16 r;
    r.q[0] = *(const uint4*)(p);
    r.q[1] = *(const uint4*)(p + 16);
    return r.v;
}

// fp32 source: 4x b128 loads + native cvt (fallback path).
static __device__ inline v16bf load_tile_g(const float* __restrict__ base,
                                           int row_stride, int k0, int lane) {
    int rc = lane & 15;
    int kb = (lane >> 4) << 3;
    const float* p = base + (size_t)rc * row_stride + k0 + kb;
    float4 f0 = *(const float4*)(p);
    float4 f1 = *(const float4*)(p + 4);
    float4 f2 = *(const float4*)(p + 16);
    float4 f3 = *(const float4*)(p + 20);
    V16 r;
    r.s[0]  = (__bf16)f0.x; r.s[1]  = (__bf16)f0.y; r.s[2]  = (__bf16)f0.z; r.s[3]  = (__bf16)f0.w;
    r.s[4]  = (__bf16)f1.x; r.s[5]  = (__bf16)f1.y; r.s[6]  = (__bf16)f1.z; r.s[7]  = (__bf16)f1.w;
    r.s[8]  = (__bf16)f2.x; r.s[9]  = (__bf16)f2.y; r.s[10] = (__bf16)f2.z; r.s[11] = (__bf16)f2.w;
    r.s[12] = (__bf16)f3.x; r.s[13] = (__bf16)f3.y; r.s[14] = (__bf16)f3.z; r.s[15] = (__bf16)f3.w;
    return r.v;
}

// LDS source (already bf16): 2x ds_load_b128.
static __device__ inline v16bf load_tile_l(const __bf16* buf,
                                           int row_stride, int k0, int lane) {
    int rc = lane & 15;
    int kb = (lane >> 4) << 3;
    const __bf16* p = buf + rc * row_stride + k0 + kb;
    V16 r;
    r.q[0] = *(const uint4*)(p);
    r.q[1] = *(const uint4*)(p + 16);
    return r.v;
}

static __device__ inline v8f wmma_bf16(v16bf a, v16bf b, v8f c) {
    return __builtin_amdgcn_wmma_f32_16x16x32_bf16(
        /*neg_a=*/false, a, /*neg_b=*/false, b,
        /*c_mod=*/(short)0, c, /*reuse_a=*/false, /*reuse_b=*/false);
}

// ---------------------------------------------------------------------------
// Kernel: elementwise fp32 -> bf16 (8 elements/thread, b128 in / b128 out)
// ---------------------------------------------------------------------------
__global__ __launch_bounds__(256)
void f32_to_bf16_kernel(const float* __restrict__ src, __bf16* __restrict__ dst, int n) {
    int i = (blockIdx.x * 256 + threadIdx.x) * 8;
    if (i >= n) return;
    float4 a = *(const float4*)(src + i);
    float4 b = *(const float4*)(src + i + 4);
    V8B r;
    r.s[0] = (__bf16)a.x; r.s[1] = (__bf16)a.y; r.s[2] = (__bf16)a.z; r.s[3] = (__bf16)a.w;
    r.s[4] = (__bf16)b.x; r.s[5] = (__bf16)b.y; r.s[6] = (__bf16)b.z; r.s[7] = (__bf16)b.w;
    *(uint4*)(dst + i) = r.q;
}

// ---------------------------------------------------------------------------
// Kernel: zero per-expert counters
// ---------------------------------------------------------------------------
__global__ void zero_counts_kernel(int* counts) {
    if (threadIdx.x < EE) counts[threadIdx.x] = 0;
}

// ---------------------------------------------------------------------------
// Kernel: router (noaux_tc grouped top-k) + scatter into expert buckets.
// One wave per token; lane e computes expert e's logit.
// ---------------------------------------------------------------------------
__global__ __launch_bounds__(32)
void route_kernel(const float* __restrict__ x,
                  const float* __restrict__ gate_w,
                  const float* __restrict__ bias,
                  int* __restrict__ counts,
                  int* __restrict__ ids,
                  float* __restrict__ wts) {
    __shared__ float sS[EE];    // unbiased sigmoid scores
    __shared__ float scS[EE];   // biased scores (selection only)
    int t = blockIdx.x;
    int e = threadIdx.x;

    const float4* xr = (const float4*)(x + (size_t)t * HH);
    const float4* gr = (const float4*)(gate_w + (size_t)e * HH);
    float acc = 0.0f;
#pragma unroll 4
    for (int i = 0; i < HH / 4; ++i) {
        float4 a = xr[i], b = gr[i];
        acc += a.x * b.x + a.y * b.y + a.z * b.z + a.w * b.w;
    }
    float score = fast_rcp(1.0f + __expf(-acc));
    sS[e] = score;
    scS[e] = score + bias[e];
    __syncthreads();

    if (e == 0) {
        const float NEG = -3.0e38f;
        float gs[NGR];
        for (int g = 0; g < NGR; ++g) {          // top-2 sums per group of 4
            float m1 = NEG, m2 = NEG;
            for (int j = 0; j < 4; ++j) {
                float v = scS[g * 4 + j];
                if (v > m1) { m2 = m1; m1 = v; }
                else if (v > m2) { m2 = v; }
            }
            gs[g] = m1 + m2;
        }
        bool gsel[NGR];
        for (int g = 0; g < NGR; ++g) gsel[g] = false;
        for (int it = 0; it < TGR; ++it) {       // top-TG groups, ties->lowest idx
            float best = NEG; int bi = 0;
            for (int g = 0; g < NGR; ++g)
                if (!gsel[g] && gs[g] > best) { best = gs[g]; bi = g; }
            gsel[bi] = true;
        }
        float cand[EE];
        for (int i = 0; i < EE; ++i) cand[i] = gsel[i >> 2] ? scS[i] : NEG;
        int   idx[KK];
        float wk[KK];
        float wsum = 0.0f;
        for (int k = 0; k < KK; ++k) {           // top-K experts in allowed groups
            float best = NEG; int bi = 0;
            for (int i = 0; i < EE; ++i)
                if (cand[i] > best) { best = cand[i]; bi = i; }
            idx[k] = bi;
            cand[bi] = -3.2e38f;
            wk[k] = sS[bi];
            wsum += wk[k];
        }
        float inv = RSCALE * fast_rcp(wsum + 1e-20f);
        for (int k = 0; k < KK; ++k) {
            int ee = idx[k];
            int pos = atomicAdd(&counts[ee], 1);
            ids[ee * TT + pos] = t;
            wts[ee * TT + pos] = wk[k] * inv;
        }
    }
}

// ---------------------------------------------------------------------------
// LDS x-staging helpers (bf16 source = straight b128 copies; f32 = convert)
// ---------------------------------------------------------------------------
static __device__ inline void stage_x(__bf16* xbuf, const __bf16* __restrict__ x,
                                      const int* tokS, int tid) {
    // 16 rows x 1024 bf16 = 16 x 128 uint4 chunks
    for (int i = tid; i < 16 * 128; i += 256) {
        int m = i >> 7;
        int c = i & 127;
        ((uint4*)xbuf)[m * 128 + c] =
            ((const uint4*)(x + (size_t)tokS[m] * HH))[c];
    }
}

static __device__ inline void stage_x(__bf16* xbuf, const float* __restrict__ x,
                                      const int* tokS, int tid) {
    for (int i = tid; i < 16 * HH; i += 256) {
        int m = i >> 10;
        int h = i & (HH - 1);
        xbuf[i] = (__bf16)x[(size_t)tokS[m] * HH + h];
    }
}

// ---------------------------------------------------------------------------
// Kernel: shared expert FFN (dense), plain stores to out.
// 256 threads = 8 wave32; one block = 16 tokens end-to-end.
// ---------------------------------------------------------------------------
template <typename WT>
__global__ __launch_bounds__(256)
void shared_expert_kernel(const WT* __restrict__ x,
                          const WT* __restrict__ sgu,   // [2*FSH, HH]
                          const WT* __restrict__ sdn,   // [HH, FSH]
                          float* __restrict__ out) {
    __shared__ __bf16 hbuf[16 * FSH];   // 32 KB bf16 activations
    int t0 = blockIdx.x * 16;
    int tid = threadIdx.x;
    int lane = tid & 31;
    int w = tid >> 5;

    const WT* xbase = x + (size_t)t0 * HH;

    // Phase 1: h = silu(x @ Wg^T) * (x @ Wu^T)   -> 64 N-tiles, 8 per wave
    for (int hv = 0; hv < 8; ++hv) {
        int nt = w * 8 + hv;                    // 0..63 (FSH/16)
        v8f ag = {}, au = {};
        const WT* bg = sgu + (size_t)(nt * 16) * HH;
        const WT* bu = sgu + (size_t)(FSH + nt * 16) * HH;
        for (int ks = 0; ks < HH / 32; ++ks) {
            int k0 = ks * 32;
            v16bf A  = load_tile_g(xbase, HH, k0, lane);
            v16bf Bg = load_tile_g(bg,    HH, k0, lane);
            v16bf Bu = load_tile_g(bu,    HH, k0, lane);
            ag = wmma_bf16(A, Bg, ag);
            au = wmma_bf16(A, Bu, au);
        }
        int col = nt * 16 + (lane & 15);
        for (int v = 0; v < 8; ++v) {
            int m = v + ((lane >> 4) << 3);
            hbuf[m * FSH + col] = (__bf16)(silu(ag[v]) * au[v]);
        }
    }
    __syncthreads();

    // Phase 2: out = h @ Wdown^T
    for (int yv = 0; yv < 8; ++yv) {
        int nt = w * 8 + yv;                    // 0..63 (HH/16)
        v8f acc = {};
        const WT* bd = sdn + (size_t)(nt * 16) * FSH;
        for (int ks = 0; ks < FSH / 32; ++ks) {
            int k0 = ks * 32;
            v16bf A = load_tile_l(hbuf, FSH, k0, lane);
            v16bf B = load_tile_g(bd,   FSH, k0, lane);
            acc = wmma_bf16(A, B, acc);
        }
        int col = nt * 16 + (lane & 15);
        for (int v = 0; v < 8; ++v) {
            int m = v + ((lane >> 4) << 3);
            out[(size_t)(t0 + m) * HH + col] = acc[v];
        }
    }
}

// ---------------------------------------------------------------------------
// Kernel: routed experts, sparse grouped GEMM over gathered tokens.
// grid = (T/16 tile slots, E experts); early-exit on empty tiles.
// ---------------------------------------------------------------------------
template <typename WT>
__global__ __launch_bounds__(256)
void routed_expert_kernel(const WT* __restrict__ x,
                          const WT* __restrict__ wgu,   // [E, 2F, H]
                          const WT* __restrict__ wdn,   // [E, H, F]
                          const int* __restrict__ counts,
                          const int* __restrict__ ids,
                          const float* __restrict__ wts,
                          float* __restrict__ out) {
    __shared__ __bf16 xbuf[16 * HH];    // 32 KB gathered tokens
    __shared__ __bf16 hbuf[16 * FF];    // 16 KB intermediate
    __shared__ int   tokS[16];
    __shared__ float wS[16];

    int e = blockIdx.y;
    int cnt = counts[e];
    int r0 = blockIdx.x * 16;
    if (r0 >= cnt) return;                      // uniform per block

    int tid = threadIdx.x;
    int lane = tid & 31;
    int w = tid >> 5;

    if (tid < 16) {
        int r = r0 + tid;
        bool valid = r < cnt;
        tokS[tid] = valid ? ids[e * TT + r] : 0;
        wS[tid]   = valid ? wts[e * TT + r] : 0.0f;   // padded rows add 0
    }
    __syncthreads();

    stage_x(xbuf, x, tokS, tid);
    __syncthreads();

    // Phase 1: gate+up -> silu*mul -> hbuf.  32 h-tiles, 4 per wave.
    const WT* wge = wgu + (size_t)e * (2 * FF * HH);
    for (int hv = 0; hv < 4; ++hv) {
        int nt = w * 4 + hv;                    // 0..31 (FF/16)
        v8f ag = {}, au = {};
        const WT* bg = wge + (size_t)(nt * 16) * HH;
        const WT* bu = wge + (size_t)(FF + nt * 16) * HH;
        for (int ks = 0; ks < HH / 32; ++ks) {
            int k0 = ks * 32;
            v16bf A  = load_tile_l(xbuf, HH, k0, lane);
            v16bf Bg = load_tile_g(bg,   HH, k0, lane);
            v16bf Bu = load_tile_g(bu,   HH, k0, lane);
            ag = wmma_bf16(A, Bg, ag);
            au = wmma_bf16(A, Bu, au);
        }
        int col = nt * 16 + (lane & 15);
        for (int v = 0; v < 8; ++v) {
            int m = v + ((lane >> 4) << 3);
            hbuf[m * FF + col] = (__bf16)(silu(ag[v]) * au[v]);
        }
    }
    __syncthreads();

    // Phase 2: down-proj, scale by routing weight, atomic accumulate.
    const WT* wde = wdn + (size_t)e * (HH * FF);
    for (int yv = 0; yv < 8; ++yv) {
        int nt = w * 8 + yv;                    // 0..63 (HH/16)
        v8f acc = {};
        const WT* bd = wde + (size_t)(nt * 16) * FF;
        for (int ks = 0; ks < FF / 32; ++ks) {
            int k0 = ks * 32;
            v16bf A = load_tile_l(hbuf, FF, k0, lane);
            v16bf B = load_tile_g(bd,   FF, k0, lane);
            acc = wmma_bf16(A, B, acc);
        }
        int col = nt * 16 + (lane & 15);
        for (int v = 0; v < 8; ++v) {
            int m = v + ((lane >> 4) << 3);
            atomicAdd(&out[(size_t)tokS[m] * HH + col], acc[v] * wS[m]);
        }
    }
}

// ---------------------------------------------------------------------------
// Host-side launch
// ---------------------------------------------------------------------------
extern "C" void kernel_launch(void* const* d_in, const int* in_sizes, int n_in,
                              void* d_out, int out_size, void* d_ws, size_t ws_size,
                              hipStream_t stream) {
    (void)in_sizes; (void)n_in; (void)out_size;

    const float* x     = (const float*)d_in[0];   // [T,H]
    const float* gatew = (const float*)d_in[1];   // [E,H]
    const float* bias  = (const float*)d_in[2];   // [E]
    const float* wgu   = (const float*)d_in[3];   // [E,2F,H]
    const float* wdn   = (const float*)d_in[4];   // [E,H,F]
    const float* sgu   = (const float*)d_in[5];   // [2FS,H]
    const float* sdn   = (const float*)d_in[6];   // [H,FS]
    float* out = (float*)d_out;                   // [T,H]

    // Workspace layout (all regions 256B-aligned):
    //   counts | ids [E*T] | wts [E*T] | bf16 copies of x and all weights
    const size_t n_x   = (size_t)TT * HH;
    const size_t n_wgu = (size_t)EE * 2 * FF * HH;
    const size_t n_wdn = (size_t)EE * HH * FF;
    const size_t n_sgu = (size_t)2 * FSH * HH;
    const size_t n_sdn = (size_t)HH * FSH;

    const size_t off_ids = 256;
    const size_t off_wts = off_ids + (size_t)EE * TT * sizeof(int);
    const size_t off_xb  = off_wts + (size_t)EE * TT * sizeof(float);
    const size_t off_gu  = off_xb + n_x * 2;
    const size_t off_dn  = off_gu + n_wgu * 2;
    const size_t off_sg  = off_dn + n_wdn * 2;
    const size_t off_sd  = off_sg + n_sgu * 2;
    const size_t need    = off_sd + n_sdn * 2;

    char* ws = (char*)d_ws;
    int*   counts = (int*)ws;
    int*   ids    = (int*)(ws + off_ids);
    float* wts    = (float*)(ws + off_wts);

    zero_counts_kernel<<<1, 32, 0, stream>>>(counts);
    route_kernel<<<TT, 32, 0, stream>>>(x, gatew, bias, counts, ids, wts);

    dim3 rgrid(TT / 16, EE);

    if (ws_size >= need) {
        // Fast path: pre-convert everything to bf16 (fits in 192MB L2),
        // then GEMMs are pure b128-load + WMMA with no conversion VALU.
        __bf16* xb  = (__bf16*)(ws + off_xb);
        __bf16* gub = (__bf16*)(ws + off_gu);
        __bf16* dnb = (__bf16*)(ws + off_dn);
        __bf16* sgb = (__bf16*)(ws + off_sg);
        __bf16* sdb = (__bf16*)(ws + off_sd);

        auto conv = [&](const float* s, __bf16* d, size_t n) {
            int blocks = (int)((n / 8 + 255) / 256);
            f32_to_bf16_kernel<<<blocks, 256, 0, stream>>>(s, d, (int)n);
        };
        conv(x,   xb,  n_x);
        conv(wgu, gub, n_wgu);
        conv(wdn, dnb, n_wdn);
        conv(sgu, sgb, n_sgu);
        conv(sdn, sdb, n_sdn);

        shared_expert_kernel<__bf16><<<TT / 16, 256, 0, stream>>>(xb, sgb, sdb, out);
        routed_expert_kernel<__bf16><<<rgrid, 256, 0, stream>>>(xb, gub, dnb,
                                                                counts, ids, wts, out);
    } else {
        // Fallback: convert fp32 operands in-register on every tile load.
        shared_expert_kernel<float><<<TT / 16, 256, 0, stream>>>(x, sgu, sdn, out);
        routed_expert_kernel<float><<<rgrid, 256, 0, stream>>>(x, wgu, wdn,
                                                               counts, ids, wts, out);
    }
}
